// GLFormerEncoder_26663156973949
// MI455X (gfx1250) — matrix operand order repaired and checked by reading, not compiled, not run on hardware
//
#include <hip/hip_runtime.h>
#include <math.h>

// ---------------------------------------------------------------------------
// GLFormer encoder, fused per-batch-row kernel for gfx1250 (MI455X).
// All GEMMs via v_wmma_f32_16x16x32_bf16 (wave32, 16x16 tiles, K=32 steps).
// ---------------------------------------------------------------------------

#define BATCH   2048
#define KTOK    64
#define HID     256
#define FF      1024
#define NTHR    256

#define SBF     272   // bf16 row stride for [64][256] buffers (272*2B = 544B, 16B aligned)
#define SFE     136   // bf16 row stride for [64][128] feature buffer (272B, 16B aligned)

// LDS layout (bytes)
#define OFF_XF     0                       // float  [64][256]   65536
#define OFF_XNB    65536                   // ushort [64][272]   34816
#define OFF_H1B    100352                  // ushort [64][272]   34816 (also feat bf16)
#define OFF_RED    135168                  // float  [64][4][2]   2048
#define OFF_MV     137216                  // float  [64][2]       512
#define OFF_ALPHA  137728                  // float  [64][8]      2048
#define OFF_OMEGA  139776                  // float  [100]         400 -> pad 416
#define OFF_DELTA  140192                  // float  [64]          256
#define OFF_WN     140448                  // float  [8]            32
#define OFF_SB     140480                  // float  [4]            16
#define SMEM_BYTES 140496

typedef __attribute__((ext_vector_type(16))) __bf16          v16bf;
typedef __attribute__((ext_vector_type(16))) unsigned short  v16u;
typedef __attribute__((ext_vector_type(8)))  unsigned short  v8u;
typedef __attribute__((ext_vector_type(8)))  float           v8f;

__device__ __forceinline__ unsigned short f2bf(float f) {
    unsigned int x = __builtin_bit_cast(unsigned int, f);
    unsigned int r = x + 0x7fffu + ((x >> 16) & 1u);   // round-to-nearest-even
    return (unsigned short)(r >> 16);
}
__device__ __forceinline__ float bf2f(unsigned short u) {
    unsigned int x = ((unsigned int)u) << 16;
    return __builtin_bit_cast(float, x);
}
__device__ __forceinline__ float gelu_exact(float x) {
    return 0.5f * x * (1.0f + erff(x * 0.70710678118654752f));
}

// A fragment: 16x32 bf16, row-major source with `stride` (elements).
// lane<16: M=lane, elems 0-7 = K[k0..k0+7], 8-15 = K[k0+16..k0+23]
// lane>=16: M=lane-16, elems 0-7 = K[k0+8..], 8-15 = K[k0+24..]
__device__ __forceinline__ v16u load_frag_a(const unsigned short* p, int stride,
                                            int m0, int k0, int lane) {
    int half = lane >> 4, r = lane & 15;
    const unsigned short* q = p + (m0 + r) * stride + k0 + half * 8;
    v8u lo = *(const v8u*)q;
    v8u hi = *(const v8u*)(q + 16);
    v16u out;
#pragma unroll
    for (int i = 0; i < 8; ++i) { out[i] = lo[i]; out[i + 8] = hi[i]; }
    return out;
}

// B fragment: 32x16 bf16 from W^T (row = N, contiguous K), stride in elements.
// lane<16: N=lane, elems 0-7 = K[k0..k0+7], 8-15 = K[k0+16..k0+23]; lane>=16: N=lane-16, K[k0+8..], K[k0+24..]
__device__ __forceinline__ v16u load_frag_b(const unsigned short* p, int stride,
                                            int n0, int k0, int lane) {
    int half = lane >> 4, r = lane & 15;
    const unsigned short* q = p + (n0 + r) * stride + k0 + half * 8;
    v8u lo = *(const v8u*)q;
    v8u hi = *(const v8u*)(q + 16);
    v16u out;
#pragma unroll
    for (int i = 0; i < 8; ++i) { out[i] = lo[i]; out[i + 8] = hi[i]; }
    return out;
}

__device__ __forceinline__ v8f wmma_bf16(v16u a, v16u b, v8f c) {
    return __builtin_amdgcn_wmma_f32_16x16x32_bf16(
        false, __builtin_bit_cast(v16bf, a),
        false, __builtin_bit_cast(v16bf, b),
        (short)0, c, false, false);
}

struct Params {
    const float* delta;      // [B,64]
    const int*   lengths;    // [B]
    const float* edge;       // [B,64,2]
    const float* feat_b;     // [256]
    const unsigned short* featWT;  // [256][128] bf16
    const float* tn_g[3]; const float* tn_b[3];
    const float* wmix[3]; const float* betap[3];
    const float* cn_g[3]; const float* cn_b[3];
    const float* b1[3];   const float* b2[3];
    const unsigned short* W1T[3];  // [1024][256] bf16
    const unsigned short* W2T[3];  // [256][1024] bf16
    const float* ln_g; const float* ln_b;
    float* out;              // [B,256]
};

// per-row mean / inv-std over 256 columns of xf
__device__ __forceinline__ void ln_stats(const float* xf, float* red, float* mv, int t) {
    int r = t >> 2, q = t & 3;
    const float* row = xf + r * HID + q * 64;
    float s = 0.f, s2 = 0.f;
#pragma unroll 4
    for (int i = 0; i < 64; ++i) { float v = row[i]; s += v; s2 += v * v; }
    red[(r * 4 + q) * 2] = s; red[(r * 4 + q) * 2 + 1] = s2;
    __syncthreads();
    if (t < KTOK) {
        float ss = 0.f, ss2 = 0.f;
#pragma unroll
        for (int q2 = 0; q2 < 4; ++q2) { ss += red[(t * 4 + q2) * 2]; ss2 += red[(t * 4 + q2) * 2 + 1]; }
        float mean = ss * (1.f / 256.f);
        float var  = ss2 * (1.f / 256.f) - mean * mean;
        mv[t * 2] = mean; mv[t * 2 + 1] = rsqrtf(var + 1e-5f);
    }
    __syncthreads();
}

// normalize xf into bf16 dst with affine (g,b); each thread owns column t
__device__ __forceinline__ void ln_apply_bf(const float* xf, unsigned short* dst,
                                            const float* g, const float* bv,
                                            const float* mv, int t) {
    float gt = g[t], bt = bv[t];
    for (int k = 0; k < KTOK; ++k) {
        float v = (xf[k * HID + t] - mv[k * 2]) * mv[k * 2 + 1] * gt + bt;
        dst[k * SBF + t] = f2bf(v);
    }
}

__global__ __launch_bounds__(NTHR)
void glformer_kernel(Params P) {
    extern __shared__ char smem[];
    float*          xf    = (float*)(smem + OFF_XF);
    unsigned short* xnb   = (unsigned short*)(smem + OFF_XNB);
    unsigned short* h1b   = (unsigned short*)(smem + OFF_H1B);
    unsigned short* feb   = h1b;                     // feature bf16 reuses h1b
    float*          red   = (float*)(smem + OFF_RED);
    float*          mv    = (float*)(smem + OFF_MV);
    float*          alph  = (float*)(smem + OFF_ALPHA);
    float*          omg   = (float*)(smem + OFF_OMEGA);
    float*          dls   = (float*)(smem + OFF_DELTA);
    float*          wn    = (float*)(smem + OFF_WN);
    float*          sb    = (float*)(smem + OFF_SB);

    const int b    = blockIdx.x;
    const int t    = threadIdx.x;
    const int lane = t & 31;
    const int wave = t >> 5;
    const int len  = P.lengths[b];
    const int lenc = len < KTOK ? len : KTOK;

    // ---- stage 0: omega, delta, bf16 feature matrix [64][128] -------------
    if (t < 100) omg[t] = __expf((float)t * (-9.0f / 99.0f) * 2.302585093f);
    if (t < KTOK) dls[t] = P.delta[b * KTOK + t];
    __syncthreads();
    for (int idx = t; idx < KTOK * 128; idx += NTHR) {
        int k = idx >> 7, c = idx & 127;
        float v;
        if (c < 100)      v = __cosf(dls[k] * omg[c]);
        else if (c < 102) v = P.edge[(b * KTOK + k) * 2 + (c - 100)];
        else              v = 0.f;
        feb[k * SFE + c] = f2bf(v);
    }
    __syncthreads();

    // ---- feature GEMM: x = feat @ feat_W + feat_b, masked -----------------
#pragma unroll
    for (int ni2 = 0; ni2 < 2; ++ni2) {
        const int n0 = (wave * 2 + ni2) * 16;
#pragma unroll
        for (int mi = 0; mi < 4; ++mi) {
            v8f acc = {};
#pragma unroll
            for (int ks = 0; ks < 4; ++ks) {
                v16u a  = load_frag_a(feb, SFE, mi * 16, ks * 32, lane);
                v16u bb = load_frag_b(P.featWT, 128, n0, ks * 32, lane);
                acc = wmma_bf16(a, bb, acc);
            }
            const int n = lane & 15, mg = (lane >> 4) * 8;
            const float bias = P.feat_b[n0 + n];
#pragma unroll
            for (int r = 0; r < 8; ++r) {
                int m = mi * 16 + mg + r;
                float v = acc[r] + bias;
                xf[m * HID + n0 + n] = (m < lenc) ? v : 0.f;
            }
        }
    }
    __syncthreads();

    // ---- 3 layers ----------------------------------------------------------
    for (int l = 0; l < 3; ++l) {
        const int cnt = (l < 2) ? 3 : 5;
        const int pstart = 2 * l;

        if (t == 0) {  // softmax(w), sigmoid(beta)
            float wv[5], mx = -1e30f;
            for (int i = 0; i < cnt; ++i) { wv[i] = P.wmix[l][i]; mx = fmaxf(mx, wv[i]); }
            float s = 0.f;
            for (int i = 0; i < cnt; ++i) { wv[i] = __expf(wv[i] - mx); s += wv[i]; }
            for (int i = 0; i < cnt; ++i) wn[i] = wv[i] / s;
            sb[0] = 1.0f / (1.0f + __expf(-P.betap[l][0]));
        }

        // LN #1 -> xnb (bf16)
        ln_stats(xf, red, mv, t);
        ln_apply_bf(xf, xnb, P.tn_g[l], P.tn_b[l], mv, t);

        // mixer blend weights per token
        if (t < KTOK) {
            const int k = t;
            const float dk = dls[k];
            const bool vmk = k < lenc;
            float td[5]; bool cv[5];
            for (int i = 0; i < cnt; ++i) {
                int p = pstart + i, kk = k + p;
                if (p == 0)           { cv[i] = vmk; td[i] = 0.f; }
                else if (kk < KTOK)   { cv[i] = vmk && (kk < lenc); td[i] = fmaxf(dls[kk] - dk, 0.f); }
                else                  { cv[i] = false; td[i] = 0.f; }
            }
            float vmax = -1e30f;
            for (int i = 0; i < cnt; ++i) { float v = cv[i] ? -td[i] : -1e9f; td[i] = v; vmax = fmaxf(vmax, v); }
            float es = 0.f;
            for (int i = 0; i < cnt; ++i) { td[i] = __expf(td[i] - vmax); es += td[i]; }
            const float beta = sb[0];
            float asum = 0.f, av[5];
            for (int i = 0; i < cnt; ++i) {
                float theta = td[i] / es;
                av[i] = cv[i] ? (beta * wn[i] + (1.f - beta) * theta) : 0.f;
                asum += av[i];
            }
            float inv = 1.0f / fmaxf(asum, 1e-8f);
            for (int i = 0; i < cnt; ++i) alph[k * 8 + i] = av[i] * inv;
        }
        __syncthreads();

        // mixer apply: xf[k][t] += sum_p alpha * xn[k+p][t]  (column t per thread)
        for (int k = 0; k < lenc; ++k) {
            float a0 = 0.f;
            for (int i = 0; i < cnt; ++i) {
                int kk = k + pstart + i;
                if (kk < KTOK) {
                    float al = alph[k * 8 + i];
                    a0 += al * bf2f(xnb[kk * SBF + t]);
                }
            }
            xf[k * HID + t] += a0;
        }
        __syncthreads();

        // LN #2 -> xnb (bf16) for FFN
        ln_stats(xf, red, mv, t);
        ln_apply_bf(xf, xnb, P.cn_g[l], P.cn_b[l], mv, t);
        __syncthreads();

        // ---- FFN, chunked over FF in 4 slices of 256 ----------------------
        v8f acc2[8];
#pragma unroll
        for (int i = 0; i < 8; ++i) acc2[i] = (v8f){};

        const unsigned short* W1T = P.W1T[l];
        const unsigned short* W2T = P.W2T[l];
        for (int c = 0; c < 4; ++c) {
            // GEMM1 slice: h1 = gelu(xn @ W1[:, c*256 : c*256+256] + b1)
#pragma unroll
            for (int ni2 = 0; ni2 < 2; ++ni2) {
                const int n0l = (wave * 2 + ni2) * 16;
                const int nG  = c * 256 + n0l;
#pragma unroll
                for (int mi = 0; mi < 4; ++mi) {
                    v8f acc = {};
#pragma unroll
                    for (int ks = 0; ks < 8; ++ks) {
                        v16u a  = load_frag_a(xnb, SBF, mi * 16, ks * 32, lane);
                        v16u bb = load_frag_b(W1T, 256, nG, ks * 32, lane);
                        acc = wmma_bf16(a, bb, acc);
                    }
                    const int n = lane & 15, mg = (lane >> 4) * 8;
                    const float bias = P.b1[l][nG + n];
#pragma unroll
                    for (int r = 0; r < 8; ++r) {
                        float v = gelu_exact(acc[r] + bias);
                        h1b[(mi * 16 + mg + r) * SBF + n0l + n] = f2bf(v);
                    }
                }
            }
            __syncthreads();

            // GEMM2 accumulate: acc2 += h1 @ W2[c*256 : c*256+256, :]
#pragma unroll
            for (int ni2 = 0; ni2 < 2; ++ni2) {
                const int n0 = (wave * 2 + ni2) * 16;
#pragma unroll
                for (int mi = 0; mi < 4; ++mi) {
                    const int idx = ni2 * 4 + mi;
#pragma unroll
                    for (int ks = 0; ks < 8; ++ks) {
                        v16u a  = load_frag_a(h1b, SBF, mi * 16, ks * 32, lane);
                        v16u bb = load_frag_b(W2T, 1024, n0, c * 256 + ks * 32, lane);
                        acc2[idx] = wmma_bf16(a, bb, acc2[idx]);
                    }
                }
            }
            __syncthreads();
        }

        // residual add: x += ffn_out + b2
#pragma unroll
        for (int ni2 = 0; ni2 < 2; ++ni2) {
            const int n0 = (wave * 2 + ni2) * 16;
#pragma unroll
            for (int mi = 0; mi < 4; ++mi) {
                const int idx = ni2 * 4 + mi;
                const int n = lane & 15, mg = (lane >> 4) * 8;
                const float bias = P.b2[l][n0 + n];
#pragma unroll
                for (int r = 0; r < 8; ++r) {
                    int m = mi * 16 + mg + r;
                    xf[m * HID + n0 + n] += acc2[idx][r] + bias;
                }
            }
        }
        __syncthreads();
    }

    // ---- final LN + masked mean pool --------------------------------------
    ln_stats(xf, red, mv, t);
    {
        const float gt = P.ln_g[t], bt = P.ln_b[t];
        float s = 0.f;
        for (int k = 0; k < lenc; ++k)
            s += (xf[k * HID + t] - mv[k * 2]) * mv[k * 2 + 1] * gt + bt;
        P.out[b * HID + t] = s / fmaxf((float)len, 1.0f);
    }
}

// ---------------------------------------------------------------------------
// Weight prep: f32 -> bf16, transposed so B-fragments are unit-stride in K.
// ---------------------------------------------------------------------------
__global__ void prep_feat_kernel(const float* __restrict__ fw, unsigned short* __restrict__ out) {
    int idx = blockIdx.x * 256 + threadIdx.x;         // 256*128
    if (idx >= 256 * 128) return;
    int n = idx >> 7, k = idx & 127;
    float v = (k < 102) ? fw[k * 256 + n] : 0.f;
    out[idx] = f2bf(v);
}

__global__ void prep_layer_kernel(const float* __restrict__ W1, const float* __restrict__ W2,
                                  unsigned short* __restrict__ W1T, unsigned short* __restrict__ W2T) {
    int idx = blockIdx.x * 256 + threadIdx.x;         // 2 * 262144
    if (idx < 262144) {
        int f = idx >> 8, k = idx & 255;              // W1T[1024][256] <- W1[256][1024]
        W1T[idx] = f2bf(W1[k * 1024 + f]);
    } else {
        int j = idx - 262144;
        int n = j >> 10, k = j & 1023;                // W2T[256][1024] <- W2[1024][256]
        W2T[j] = f2bf(W2[k * 256 + n]);
    }
}

// ---------------------------------------------------------------------------
extern "C" void kernel_launch(void* const* d_in, const int* in_sizes, int n_in,
                              void* d_out, int out_size, void* d_ws, size_t ws_size,
                              hipStream_t stream) {
    (void)in_sizes; (void)n_in; (void)out_size; (void)ws_size;

    const float* delta   = (const float*)d_in[0];
    const int*   lengths = (const int*)d_in[1];
    const float* edge    = (const float*)d_in[2];
    const float* feat_W  = (const float*)d_in[3];
    const float* feat_b  = (const float*)d_in[4];
    const float* ln_g    = (const float*)d_in[35];
    const float* ln_b    = (const float*)d_in[36];

    unsigned short* ws = (unsigned short*)d_ws;
    unsigned short* featWT = ws;                      // 32768 elements
    Params P;
    P.delta = delta; P.lengths = lengths; P.edge = edge;
    P.feat_b = feat_b; P.featWT = featWT;
    P.ln_g = ln_g; P.ln_b = ln_b; P.out = (float*)d_out;

    prep_feat_kernel<<<128, 256, 0, stream>>>(feat_W, featWT);

    for (int l = 0; l < 3; ++l) {
        const int base = 5 + 10 * l;
        P.tn_g[l]  = (const float*)d_in[base + 0];
        P.tn_b[l]  = (const float*)d_in[base + 1];
        P.wmix[l]  = (const float*)d_in[base + 2];
        P.betap[l] = (const float*)d_in[base + 3];
        P.cn_g[l]  = (const float*)d_in[base + 4];
        P.cn_b[l]  = (const float*)d_in[base + 5];
        const float* W1 = (const float*)d_in[base + 6];
        P.b1[l]    = (const float*)d_in[base + 7];
        const float* W2 = (const float*)d_in[base + 8];
        P.b2[l]    = (const float*)d_in[base + 9];

        unsigned short* w1t = ws + 32768 + (size_t)l * 524288;
        unsigned short* w2t = w1t + 262144;
        P.W1T[l] = w1t; P.W2T[l] = w2t;
        prep_layer_kernel<<<2048, 256, 0, stream>>>(W1, W2, w1t, w2t);
    }

    glformer_kernel<<<BATCH, NTHR, SMEM_BYTES, stream>>>(P);
}
